// ROIPooling_17446157157176
// MI455X (gfx1250) — matrix operand order repaired and checked by reading, not compiled, not run on hardware
//
#include <hip/hip_runtime.h>
#include <hip/hip_bf16.h>

// ---------------------------------------------------------------------------
// ROI pooling + FPN positional encoding for MI455X (gfx1250, wave32).
//   out = [pool_out (2*128*256*49 f32) | pe_out (same) | lvl (256 f32)]
// Memory-bound (~200MB @ 23.3TB/s). Pooling = per-ROI (49 x K) x (K x 256)
// GEMM on v_wmma_f32_16x16x32_f16 with an exact 0/1 indicator A-matrix built
// once per K-superchunk in LDS; A/B fragments are ds_load_b128 pairs.
// PE table rows gathered via TDM gather-mode tensor_load_to_lds (6-arg form).
// ---------------------------------------------------------------------------

typedef _Float16 v16h __attribute__((ext_vector_type(16)));
typedef float    v8f  __attribute__((ext_vector_type(8)));
typedef unsigned int u32x4 __attribute__((ext_vector_type(4)));
typedef int          i32x4 __attribute__((ext_vector_type(4)));
typedef int          i32x8 __attribute__((ext_vector_type(8)));

#define NLAYERS 4
#define POOLP   7
#define CFPN    256
#define BB      2
#define NN      128
#define NROI    (BB*NN)
#define HALF    128
#define TBL_ROWS 513                    // padded cumsum of sinusoidal PE
#define POOL_ELEMS ((size_t)NROI*CFPN*POOLP*POOLP)

// geometry record: 64 ints per ROI
//  0:lvl 1:y1 2:x1 3:Rs 4:Cs 5:Hl 6:Wl
//  8..14 r_lo  15..21 r_hi  22..28 c_lo  29..35 c_hi
//  36..42 f_lo 43..49 f_hi  50..56 t_lo  57..63 t_hi
#define GSTRIDE 64

#if defined(__has_builtin)
# if __has_builtin(__builtin_amdgcn_tensor_load_to_lds)
#  define USE_TDM 1
# endif
#endif
#ifndef USE_TDM
# define USE_TDM 0
#endif

// ---------------------------------------------------------------- geometry --
__global__ void geom_kernel(const float* __restrict__ rois,
                            int* __restrict__ geom,
                            float* __restrict__ lvl_out) {
  int t = blockIdx.x * blockDim.x + threadIdx.x;
  if (t >= NROI) return;
  float x1f = rois[t*4+0], y1f = rois[t*4+1], x2f = rois[t*4+2], y2f = rois[t*4+3];
  float side = sqrtf((x2f - x1f) * (y2f - y1f));
  int lvl = (int)truncf(log2f(side * 0.1f));
  lvl = min(max(lvl, 0), NLAYERS - 1);
  float strf = (float)(2 << lvl);                 // 2^(lvl+1)
  int Hl = 512 >> (lvl + 1), Wl = Hl;
  int x1 = __float2int_rn(x1f / strf);
  int y1 = __float2int_rn(y1f / strf);
  int x2 = __float2int_rn(x2f / strf);
  int y2 = __float2int_rn(y2f / strf);
  y2 = min(y2, Hl - 1);
  #pragma unroll
  for (int it = 0; it < POOLP; ++it)
    if (y2 - y1 + 1 < POOLP) { y1 = max(y1 - 1, 0); y2 = min(y2 + 1, Hl - 1); }
  #pragma unroll
  for (int it = 0; it < POOLP; ++it)
    if (x2 - x1 + 1 < POOLP) { x1 = max(x1 - 1, 0); x2 = min(x2 + 1, Wl - 1); }
  int x2c = min(x2, Wl - 1);
  int Rs = y2 - y1 + 1, Cs = x2c - x1 + 1;

  int* g = geom + (size_t)t * GSTRIDE;
  g[0]=lvl; g[1]=y1; g[2]=x1; g[3]=Rs; g[4]=Cs; g[5]=Hl; g[6]=Wl; g[7]=0;
  int st = 2 << lvl;
  int fS = st * (y2 - y1), f0 = st * y1;
  int tS = min(st * (x2 - x1), 512);
  #pragma unroll
  for (int i = 0; i < POOLP; ++i) {
    g[8 +i] = y1 + (i*Rs)/POOLP;  g[15+i] = y1 + ((i+1)*Rs + POOLP-1)/POOLP;
    g[22+i] = x1 + (i*Cs)/POOLP;  g[29+i] = x1 + ((i+1)*Cs + POOLP-1)/POOLP;
    g[36+i] = f0 + (i*fS)/POOLP;  g[43+i] = f0 + ((i+1)*fS + POOLP-1)/POOLP;
    g[50+i] =      (i*tS)/POOLP;  g[57+i] =      ((i+1)*tS + POOLP-1)/POOLP;
  }
  lvl_out[t] = (float)lvl;
}

// ----------------------------------------------- sinusoidal-PE cumsum table --
__global__ void pe_table_kernel(float* __restrict__ table) {
  int c = threadIdx.x;                       // 0..127
  float freq = expf(-(logf(10000.f) / (float)HALF) * (float)(c & ~1));
  bool isCos = (c & 1);
  float acc = 0.f;
  table[c] = 0.f;
  for (int p = 0; p < 512; ++p) {
    float ang = (float)p * freq;
    acc += isCos ? cosf(ang) : sinf(ang);
    table[(size_t)(p + 1) * HALF + c] = acc;
  }
}

// ------------------------------------------------------ WMMA pooling kernel --
// 1 block = 1 ROI, 256 threads = 8 waves. pooled = Ind(49xK) * Patch(Kx256).
#define KCH  256                      // staged K superchunk
#define KROW 264                      // padded LDS row (halves): 528B, 16B-mult

__global__ void __launch_bounds__(256)
pool_wmma_kernel(const float* __restrict__ fm0, const float* __restrict__ fm1,
                 const float* __restrict__ fm2, const float* __restrict__ fm3,
                 const int* __restrict__ geom, float* __restrict__ pool_out) {
  __shared__ __align__(16) _Float16 patch[64][KROW];   // f16 fm slice (ct,ks)
  __shared__ __align__(16) _Float16 sInd[64][KROW];    // 0/1 indicator (ks)
  __shared__ int krc[1024];                            // abs (r<<16)|c per px
  __shared__ int sRlo[POOLP], sRhi[POOLP], sClo[POOLP], sChi[POOLP];
  __shared__ float sArea[49];

  const int roi = blockIdx.x;
  const int tid = threadIdx.x, wid = tid >> 5, ln = tid & 31;
  const int l16 = ln & 15; const bool lnHi = ln >= 16;
  const int* g = geom + (size_t)roi * GSTRIDE;
  const int lvl = g[0], y1 = g[1], x1 = g[2], Rs = g[3], Cs = g[4];
  const int Wl = g[6], Hl = g[5];
  const int bidx = roi >> 7;                            // roi / N
  const float* fm = (lvl == 0) ? fm0 : (lvl == 1) ? fm1 : (lvl == 2) ? fm2 : fm3;

  if (tid < POOLP) { sRlo[tid]=g[8+tid]; sRhi[tid]=g[15+tid];
                     sClo[tid]=g[22+tid]; sChi[tid]=g[29+tid]; }
  const int K = Rs * Cs;                                // <= 1024
  int Kpad = (K + 31) & ~31; if (Kpad > 1024) Kpad = 1024;
  for (int k = tid; k < 1024; k += 256) {               // sentinel fails bounds
    int v = 0x7FFF7FFF;
    if (k < K) { int r = k / Cs, c = k - r * Cs; v = ((y1+r) << 16) | (x1+c); }
    krc[k] = v;
  }
  __syncthreads();
  if (tid < 49)
    sArea[tid] = (float)((sRhi[tid/7]-sRlo[tid/7]) * (sChi[tid%7]-sClo[tid%7]));

  // wave's two tiles: mts differ, nt shared -> B fragment reused by both WMMAs
  const int mt0 = (wid * 2) & 3, mt1 = (wid * 2 + 1) & 3;
  const int nt  = wid >> 1;
  const size_t plane = (size_t)Hl * Wl;

  v8f acc[4][2];
  #pragma unroll
  for (int ct = 0; ct < 4; ++ct) { acc[ct][0] = (v8f){}; acc[ct][1] = (v8f){}; }

  for (int ks = 0; ks < Kpad; ks += KCH) {
    const int kEff = min(KCH, Kpad - ks);               // multiple of 32
    __syncthreads();                                    // sInd/patch reuse fence
    // ---- build 64x256 indicator chunk: thread = pixel column, separable masks
    {
      const int rc = krc[ks + tid];
      const int r = rc >> 16, c = rc & 0xFFFF;
      unsigned rowM = 0, colM = 0;
      #pragma unroll
      for (int i = 0; i < POOLP; ++i) {
        rowM |= (unsigned)(r >= sRlo[i] && r < sRhi[i]) << i;
        colM |= (unsigned)(c >= sClo[i] && c < sChi[i]) << i;
      }
      _Float16* col = &sInd[0][0] + tid;
      #pragma unroll
      for (int m = 0; m < 64; ++m) {
        const bool on = (m < 49) && ((rowM >> (m / 7)) & 1u)
                                 && ((colM >> (m % 7)) & 1u);
        col[m * KROW] = on ? (_Float16)1.f : (_Float16)0.f;
      }
    }
    #pragma unroll
    for (int ct = 0; ct < 4; ++ct) {                    // 64-channel tiles
      __syncthreads();                                  // ind ready / patch free
      // ---- stage f16 patch chunk: 8 channels per wave, k coalesced per lane
      const size_t fbase = ((size_t)bidx * CFPN + ct * 64) * plane;
      for (int ci = 0; ci < 8; ++ci) {
        const int cc = wid * 8 + ci;
        const size_t cbase = fbase + (size_t)cc * plane;
        for (int k = ln; k < kEff; k += 32) {
          const int kg = ks + k;
          float v = 0.f;
          if (kg < K) {
            const int rc = krc[kg];
            v = fm[cbase + (size_t)(rc >> 16) * Wl + (rc & 0xFFFF)];
          }
          patch[cc][k] = (_Float16)v;
        }
        if (ks + KCH < Kpad && ln == 0)                 // hint next superchunk
          __builtin_prefetch(&fm[cbase + (size_t)y1 * Wl + x1], 0, 1);
      }
      __syncthreads();
      // ---- GEMM over 32-pixel chunks: all fragment loads are ds_load_b128
      const int nkc = kEff >> 5;
      const char* brow = (const char*)&patch[nt * 16 + l16][0] + (lnHi ? 16 : 0);
      const char* ar0  = (const char*)&sInd[mt0 * 16 + l16][0] + (lnHi ? 16 : 0);
      const char* ar1  = (const char*)&sInd[mt1 * 16 + l16][0] + (lnHi ? 16 : 0);
      for (int kc = 0; kc < nkc; ++kc) {
        union { v16h h; uint4 q[2]; } bf, a0, a1;
        bf.q[0] = *(const uint4*)(brow + kc * 64);
        bf.q[1] = *(const uint4*)(brow + kc * 64 + 32);
        a0.q[0] = *(const uint4*)(ar0 + kc * 64);
        a0.q[1] = *(const uint4*)(ar0 + kc * 64 + 32);
        a1.q[0] = *(const uint4*)(ar1 + kc * 64);
        a1.q[1] = *(const uint4*)(ar1 + kc * 64 + 32);
        acc[ct][0] = __builtin_amdgcn_wmma_f32_16x16x32_f16(
            false, a0.h, false, bf.h, (short)0, acc[ct][0], false, false);
        acc[ct][1] = __builtin_amdgcn_wmma_f32_16x16x32_f16(
            false, a1.h, false, bf.h, (short)0, acc[ct][1], false, false);
      }
    }
  }
  // ---- write out (ISA 16x16 f32 D layout), divide by bin area
  #pragma unroll
  for (int ct = 0; ct < 4; ++ct) {
    #pragma unroll
    for (int p = 0; p < 2; ++p) {
      const int mt = p ? mt1 : mt0;
      const int ch = ct * 64 + nt * 16 + l16;
      #pragma unroll
      for (int v = 0; v < 8; ++v) {
        const int m = mt * 16 + v + (lnHi ? 8 : 0);
        if (m < 49) {
          const int bi = m / 7, bj = m - 7 * bi;
          pool_out[((size_t)roi * CFPN + ch) * 49 + bi * 7 + bj] =
              acc[ct][p][v] / sArea[m];
        }
      }
    }
  }
}

// --------------------------------------------------------- PE gather kernel --
__device__ __forceinline__ unsigned lds_byte_off(void* p) {
  return (unsigned)(unsigned long long)(__attribute__((address_space(3))) char*)p;
}

__global__ void __launch_bounds__(256)
pe_kernel(const float* __restrict__ table, const int* __restrict__ geom,
          float* __restrict__ pe_out) {
  __shared__ __align__(16) float gathF[16 * HALF];
  __shared__ __align__(16) float gathT[16 * HALF];
  __shared__ float avgF[POOLP * HALF], avgT[POOLP * HALF];
  const int roi = blockIdx.x, tid = threadIdx.x;
  const int* g = geom + (size_t)roi * GSTRIDE;

#if USE_TDM
  if (tid < 32) {
    // TDM gather-mode descriptor: 14 rows of 128 f32 from the cumsum table.
    const unsigned long long ga = (unsigned long long)table;
    u32x4 g0;
    g0[0] = 1u | (1u << 31);                 // count=1, 16-bit idx, gather_mode
    g0[1] = lds_byte_off(gathF);
    g0[2] = (unsigned)(ga & 0xFFFFFFFFu);
    g0[3] = (unsigned)((ga >> 32) & 0x01FFFFFFu) | (2u << 30);  // type=2
    i32x8 g1;
    g1[0] = (int)(2u << 16);                 // data_size = 4B
    g1[1] = (int)((unsigned)HALF << 16);     // tensor_dim0 lo16
    g1[2] = (int)(((unsigned)TBL_ROWS & 0xFFFFu) << 16); // dim0 hi=0 | dim1 lo16
    g1[3] = (int)((unsigned)HALF << 16);     // dim1 hi=0 | tile_dim0=128
    g1[4] = 14;                              // tile_dim1 = #valid row indices
    g1[5] = HALF;                            // tensor_dim0_stride
    g1[6] = 0; g1[7] = 0;
    const i32x8 gx = {};                     // extra SGPR group (unused): zeros
    int idx[16];
    #pragma unroll
    for (int q = 0; q < 7; ++q) { idx[q] = g[36 + q]; idx[7 + q] = g[43 + q]; }
    idx[14] = idx[15] = 0;
    i32x4 g2, g3;
    #pragma unroll
    for (int w = 0; w < 4; ++w) {
      g2[w] = (idx[2*w]   & 0xFFFF) | (idx[2*w+1]   << 16);
      g3[w] = (idx[8+2*w] & 0xFFFF) | (idx[8+2*w+1] << 16);
    }
    __builtin_amdgcn_tensor_load_to_lds(g0, g1, g2, g3, gx, 0);
    // second gather: column-PE rows
    g0[1] = lds_byte_off(gathT);
    #pragma unroll
    for (int q = 0; q < 7; ++q) { idx[q] = g[50 + q]; idx[7 + q] = g[57 + q]; }
    #pragma unroll
    for (int w = 0; w < 4; ++w) {
      g2[w] = (idx[2*w]   & 0xFFFF) | (idx[2*w+1]   << 16);
      g3[w] = (idx[8+2*w] & 0xFFFF) | (idx[8+2*w+1] << 16);
    }
    __builtin_amdgcn_tensor_load_to_lds(g0, g1, g2, g3, gx, 0);
  }
# if __has_builtin(__builtin_amdgcn_s_wait_tensorcnt)
  __builtin_amdgcn_s_wait_tensorcnt(0);
# else
  asm volatile("s_wait_tensorcnt 0x0" ::: "memory");
# endif
  __syncthreads();
#else
  for (int idx = tid; idx < 14 * HALF; idx += 256) {
    const int q = idx >> 7, c = idx & (HALF - 1);
    gathF[idx] = table[(size_t)((q < 7) ? g[36+q] : g[43+q-7]) * HALF + c];
    gathT[idx] = table[(size_t)((q < 7) ? g[50+q] : g[57+q-7]) * HALF + c];
  }
  __syncthreads();
#endif

  for (int idx = tid; idx < POOLP * HALF; idx += 256) {
    const int i = idx >> 7, c = idx & (HALF - 1);
    avgF[idx] = (gathF[(7+i)*HALF+c] - gathF[i*HALF+c]) / (float)(g[43+i]-g[36+i]);
    avgT[idx] = (gathT[(7+i)*HALF+c] - gathT[i*HALF+c]) / (float)(g[57+i]-g[50+i]);
  }
  __syncthreads();
  float* out = pe_out + (size_t)roi * CFPN * 49;
  for (int idx = tid; idx < CFPN * 49; idx += 256) {
    const int c = idx / 49, ij = idx - c * 49;
    const int i = ij / 7, j = ij - i * 7;
    out[idx] = (c < HALF) ? avgF[i * HALF + c] : avgT[j * HALF + (c - HALF)];
  }
}

// ------------------------------------------------------------------ launch --
extern "C" void kernel_launch(void* const* d_in, const int* in_sizes, int n_in,
                              void* d_out, int out_size, void* d_ws, size_t ws_size,
                              hipStream_t stream) {
  (void)in_sizes; (void)n_in; (void)out_size; (void)ws_size;
  const float* rois = (const float*)d_in[0];
  const float* fm0  = (const float*)d_in[1];
  const float* fm1  = (const float*)d_in[2];
  const float* fm2  = (const float*)d_in[3];
  const float* fm3  = (const float*)d_in[4];

  float* table = (float*)d_ws;                                   // 513*128 f32
  int*   geom  = (int*)((char*)d_ws + (size_t)TBL_ROWS*HALF*4);  // 256*64 i32

  float* pool_out = (float*)d_out;
  float* pe_out   = pool_out + POOL_ELEMS;
  float* lvl_out  = pe_out + POOL_ELEMS;

  geom_kernel<<<1, 256, 0, stream>>>(rois, geom, lvl_out);
  pe_table_kernel<<<1, HALF, 0, stream>>>(table);
  pool_wmma_kernel<<<NROI, 256, 0, stream>>>(fm0, fm1, fm2, fm3, geom, pool_out);
  pe_kernel<<<NROI, 256, 0, stream>>>(table, geom, pe_out);
}
